// CausalSelfAttention_5652176961679
// MI455X (gfx1250) — matrix operand order repaired and checked by reading, not compiled, not run on hardware
//
#include <hip/hip_runtime.h>

// ---------------- problem constants ----------------
#define BB   2
#define TT   2048
#define CC   2048
#define HH   16
#define HKV  4
#define DD   128
#define RR   64
#define NQKV 3072          // C + 2*HKV*D = 2048 + 512 + 512

typedef __attribute__((ext_vector_type(16))) __bf16 v16bf;
typedef __attribute__((ext_vector_type(8)))  float  v8f;

union Frag {
    v16bf v;
    uint4 u[2];
};

__device__ inline v8f v8f_zero() {
    v8f z = {0.f,0.f,0.f,0.f,0.f,0.f,0.f,0.f};
    return z;
}

// A-fragment (16x32 bf16, MxK): rowptr points at (row m, k = kbase).
// lane<16 holds K 0..15; lane>=16 holds K 16..31 (2 per VGPR).
__device__ inline v16bf load_a16x32(const __bf16* rowptr, int hi) {
    Frag f;
    f.u[0] = *(const uint4*)(rowptr + hi * 16);
    f.u[1] = *(const uint4*)(rowptr + hi * 16 + 8);
    return f.v;
}

// B-fragment (32x16 bf16, KxN) from an N-major [N,K] buffer:
// per lane: 16 contiguous bf16 (lane<16 -> K 0..15, lane>=16 -> K 16..31).
__device__ inline v16bf load_contig16(const __bf16* p) {
    Frag f;
    f.u[0] = ((const uint4*)p)[0];
    f.u[1] = ((const uint4*)p)[1];
    return f.v;
}

__device__ inline v8f wmma_bf16(v16bf a, v16bf b, v8f c) {
    return __builtin_amdgcn_wmma_f32_16x16x32_bf16(false, a, false, b,
                                                   (short)0, c, false, false);
}

// ---------------- elementwise conversion / layout kernels ----------------
__global__ __launch_bounds__(256) void k_f32_to_bf16(const float* __restrict__ s,
                                                     __bf16* __restrict__ d, int n) {
    int i = blockIdx.x * 256 + threadIdx.x;
    if (i < n) d[i] = (__bf16)s[i];
}

// Build combined QKV weight, transposed to [N=3072][K=2048] bf16.
__global__ __launch_bounds__(256) void k_build_wqkvT(const float* __restrict__ Wq,
                                                     const float* __restrict__ Wk,
                                                     const float* __restrict__ Wv,
                                                     __bf16* __restrict__ wt) {
    int i = blockIdx.x * 256 + threadIdx.x;          // over 3072*2048
    int n = i >> 11;                                  // /2048
    int k = i & 2047;
    float v;
    if (n < CC)            v = Wq[k * CC + n];
    else if (n < CC + 512) v = Wk[k * 512 + (n - CC)];
    else                   v = Wv[k * 512 + (n - CC - 512)];
    wt[i] = (__bf16)v;
}

__global__ __launch_bounds__(256) void k_build_woT(const float* __restrict__ Wo,
                                                   __bf16* __restrict__ wt) {
    int i = blockIdx.x * 256 + threadIdx.x;          // over 2048*2048
    int n = i >> 11;
    int k = i & 2047;
    wt[i] = (__bf16)Wo[k * CC + n];
}

// RoPE on Q: qkv_raw [B*T, 3072] -> q [B,H,T,D]
__global__ __launch_bounds__(256) void k_rope_q(const __bf16* __restrict__ raw,
                                                const float* __restrict__ cosb,
                                                const float* __restrict__ sinb,
                                                __bf16* __restrict__ q) {
    int i = blockIdx.x * 256 + threadIdx.x;          // over B*H*T*D
    int d = i & 127;
    int t = (i >> 7) & 2047;
    int h = (i >> 18) & 15;
    int b = (i >> 22) & 1;
    int row = b * TT + t;
    int col = h * DD + d;
    float v = (float)raw[row * NQKV + col];
    float o = v;
    if (d < RR) {
        float partner = (d < RR / 2) ? -(float)raw[row * NQKV + col + RR / 2]
                                     :  (float)raw[row * NQKV + col - RR / 2];
        o = v * cosb[t * RR + d] + partner * sinb[t * RR + d];
    }
    q[i] = (__bf16)o;
}

// RoPE on K: -> k [B,HKV,T,D]
__global__ __launch_bounds__(256) void k_rope_k(const __bf16* __restrict__ raw,
                                                const float* __restrict__ cosb,
                                                const float* __restrict__ sinb,
                                                __bf16* __restrict__ kk) {
    int i = blockIdx.x * 256 + threadIdx.x;          // over B*HKV*T*D
    int d = i & 127;
    int t = (i >> 7) & 2047;
    int kh = (i >> 18) & 3;
    int b = (i >> 20) & 1;
    int row = b * TT + t;
    int col = CC + kh * DD + d;
    float v = (float)raw[row * NQKV + col];
    float o = v;
    if (d < RR) {
        float partner = (d < RR / 2) ? -(float)raw[row * NQKV + col + RR / 2]
                                     :  (float)raw[row * NQKV + col - RR / 2];
        o = v * cosb[t * RR + d] + partner * sinb[t * RR + d];
    }
    kk[i] = (__bf16)o;
}

// V transposed: vT [B,HKV,D,T]
__global__ __launch_bounds__(256) void k_transpose_v(const __bf16* __restrict__ raw,
                                                     __bf16* __restrict__ vT) {
    int i = blockIdx.x * 256 + threadIdx.x;          // over B*HKV*D*T (vT index)
    int t = i & 2047;
    int d = (i >> 11) & 127;
    int kh = (i >> 18) & 3;
    int b = (i >> 20) & 1;
    vT[i] = raw[(b * TT + t) * NQKV + (CC + 512) + kh * DD + d];
}

// ---------------- WMMA GEMM: C[M,N] = A[M,K] * Bt[N,K]^T ----------------
// Block = 8 waves (2 M x 4 N); block tile 128x128; wave tile 64x32
// (4 M-tiles x 2 N-tiles). K stepped by 32 with double-buffered LDS staging:
// As/Bs = 128x32 bf16 (8KB) each, x2 buffers = 32KB LDS.
template <int OUTF32>
__global__ __launch_bounds__(256) void k_gemm_bf16(const __bf16* __restrict__ A,
                                                   const __bf16* __restrict__ Bt,
                                                   void* __restrict__ Cout,
                                                   int M, int N, int K) {
    __shared__ __align__(16) __bf16 As[2][128 * 32];
    __shared__ __align__(16) __bf16 Bs[2][128 * 32];

    const int tid  = threadIdx.x;
    const int lane = tid & 31;
    const int wid  = tid >> 5;
    const int hi = lane >> 4, nn = lane & 15;
    const int wy = wid >> 2;          // 0..1  (M direction)
    const int wx = wid & 3;           // 0..3  (N direction)
    const int row0 = blockIdx.y * 128;
    const int col0 = blockIdx.x * 128;

    // global staging: each thread moves one 32B chunk of A and of B per K-step
    const int lrow = tid >> 1;            // 0..127
    const int lcol = (tid & 1) * 16;      // 0 or 16 (elements)
    const __bf16* aGlob = A  + (size_t)(row0 + lrow) * K + lcol;
    const __bf16* bGlob = Bt + (size_t)(col0 + lrow) * K + lcol;
    __bf16* aStage = &As[0][lrow * 32 + lcol];   // [buf] chosen via offset
    __bf16* bStage = &Bs[0][lrow * 32 + lcol];
    const int bufStride = 128 * 32;

    v8f acc[4][2];
#pragma unroll
    for (int mt = 0; mt < 4; ++mt)
#pragma unroll
        for (int nt = 0; nt < 2; ++nt) acc[mt][nt] = v8f_zero();

    const int nk = K / 32;

    // prologue: tile 0 -> buffer 0
    uint4 a0 = ((const uint4*)aGlob)[0];
    uint4 a1 = ((const uint4*)aGlob)[1];
    uint4 b0 = ((const uint4*)bGlob)[0];
    uint4 b1 = ((const uint4*)bGlob)[1];
    ((uint4*)aStage)[0] = a0; ((uint4*)aStage)[1] = a1;
    ((uint4*)bStage)[0] = b0; ((uint4*)bStage)[1] = b1;
    __syncthreads();

    for (int kb = 0; kb < nk; ++kb) {
        const int cur = kb & 1;
        const int nxt = cur ^ 1;

        // issue next tile's global loads early (overlaps with WMMAs below)
        if (kb + 1 < nk) {
            const __bf16* ap = aGlob + (kb + 1) * 32;
            const __bf16* bp = bGlob + (kb + 1) * 32;
            a0 = ((const uint4*)ap)[0];
            a1 = ((const uint4*)ap)[1];
            b0 = ((const uint4*)bp)[0];
            b1 = ((const uint4*)bp)[1];
        }

        // fragments from LDS
        v16bf af[4], bf[2];
#pragma unroll
        for (int mt = 0; mt < 4; ++mt)
            af[mt] = load_a16x32(&As[cur][(wy * 64 + mt * 16 + nn) * 32], hi);
#pragma unroll
        for (int nt = 0; nt < 2; ++nt)
            bf[nt] = load_contig16(&Bs[cur][(wx * 32 + nt * 16 + nn) * 32 + hi * 16]);

#pragma unroll
        for (int mt = 0; mt < 4; ++mt)
#pragma unroll
            for (int nt = 0; nt < 2; ++nt)
                acc[mt][nt] = wmma_bf16(af[mt], bf[nt], acc[mt][nt]);

        if (kb + 1 < nk) {
            uint4* as = (uint4*)(aStage + nxt * bufStride);
            uint4* bs = (uint4*)(bStage + nxt * bufStride);
            as[0] = a0; as[1] = a1;
            bs[0] = b0; bs[1] = b1;
        }
        __syncthreads();
    }

    // epilogue: C-tile layout -> memory
#pragma unroll
    for (int mt = 0; mt < 4; ++mt)
#pragma unroll
        for (int nt = 0; nt < 2; ++nt)
#pragma unroll
            for (int r = 0; r < 8; ++r) {
                int row = row0 + wy * 64 + mt * 16 + hi * 8 + r;
                int col = col0 + wx * 32 + nt * 16 + nn;
                if (OUTF32) ((float*)Cout)[(size_t)row * N + col] = acc[mt][nt][r];
                else        ((__bf16*)Cout)[(size_t)row * N + col] = (__bf16)acc[mt][nt][r];
            }
}

// ---------------- attention: two-pass flash with threshold gate ----------------
// One wave per 16-row query tile. Q [B,H,T,D], K [B,HKV,T,D], vT [B,HKV,D,T].
__global__ __launch_bounds__(256) void k_attn(const __bf16* __restrict__ q,
                                              const __bf16* __restrict__ kmat,
                                              const __bf16* __restrict__ vT,
                                              const float* __restrict__ gate,
                                              __bf16* __restrict__ y) {
    __shared__ __align__(16) __bf16 pbuf[8][16 * 32];   // 1KB per wave

    int lane = threadIdx.x & 31;
    int wid  = threadIdx.x >> 5;
    int hi = lane >> 4, nn = lane & 15;

    int tile = blockIdx.x * 8 + wid;
    int tq = tile & 127;            // T/16 = 128 tiles
    int h  = (tile >> 7) & 15;
    int b  = tile >> 11;
    int kh = h >> 2;                // H/HKV = 4

    const float scale = 0.08838834764831845f;   // D^-0.5
    float thr = 1.f / (1.f + __expf(-gate[h]));

    const __bf16* qb = q    + (size_t)((b * HH + h) * TT + tq * 16) * DD;
    const __bf16* kb = kmat + (size_t)((b * HKV + kh) * TT) * DD;
    const __bf16* vb = vT   + (size_t)((b * HKV + kh) * DD) * TT;

    // Q fragments: 4 chunks of K=32 along D
    v16bf qf[4];
#pragma unroll
    for (int dc = 0; dc < 4; ++dc)
        qf[dc] = load_a16x32(qb + nn * DD + dc * 32, hi);

    // ---- pass 1: row max + row sum (online) ----
    float m8[8], l8[8];
#pragma unroll
    for (int r = 0; r < 8; ++r) { m8[r] = -__builtin_inff(); l8[r] = 0.f; }

    for (int j = 0; j <= tq; ++j) {
        v8f acc = v8f_zero();
        const __bf16* kt = kb + (size_t)(j * 16 + nn) * DD + hi * 16;
#pragma unroll
        for (int dc = 0; dc < 4; ++dc)
            acc = wmma_bf16(qf[dc], load_contig16(kt + dc * 32), acc);

        float sc[8];
        bool diag = (j == tq);
#pragma unroll
        for (int r = 0; r < 8; ++r) {
            float s = acc[r] * scale;
            if (diag && nn > (r + hi * 8)) s = -__builtin_inff();
            sc[r] = s;
        }
        float rm[8];
#pragma unroll
        for (int r = 0; r < 8; ++r) rm[r] = sc[r];
        for (int o = 1; o < 16; o <<= 1)
#pragma unroll
            for (int r = 0; r < 8; ++r)
                rm[r] = fmaxf(rm[r], __shfl_xor(rm[r], o, 32));

        float p[8];
#pragma unroll
        for (int r = 0; r < 8; ++r) {
            float nm = fmaxf(m8[r], rm[r]);
            float alpha = __expf(m8[r] - nm);
            p[r] = __expf(sc[r] - nm);
            l8[r] = l8[r] * alpha;
            m8[r] = nm;
        }
        for (int o = 1; o < 16; o <<= 1)
#pragma unroll
            for (int r = 0; r < 8; ++r)
                p[r] += __shfl_xor(p[r], o, 32);
#pragma unroll
        for (int r = 0; r < 8; ++r) l8[r] += p[r];
    }

    float rl[8];
#pragma unroll
    for (int r = 0; r < 8; ++r) rl[r] = 1.f / l8[r];

    // ---- pass 2: gated probabilities, att @ V ----
    v8f od[8];
#pragma unroll
    for (int dc = 0; dc < 8; ++dc) od[dc] = v8f_zero();

    int njc = (tq >> 1) + 1;
    for (int jc = 0; jc < njc; ++jc) {
#pragma unroll
        for (int jt = 0; jt < 2; ++jt) {
            int j = jc * 2 + jt;
            float pv[8];
            if (j <= tq) {
                v8f acc = v8f_zero();
                const __bf16* kt = kb + (size_t)(j * 16 + nn) * DD + hi * 16;
#pragma unroll
                for (int dc = 0; dc < 4; ++dc)
                    acc = wmma_bf16(qf[dc], load_contig16(kt + dc * 32), acc);
                bool diag = (j == tq);
#pragma unroll
                for (int r = 0; r < 8; ++r) {
                    float s = acc[r] * scale;
                    if (diag && nn > (r + hi * 8)) s = -__builtin_inff();
                    float p = __expf(s - m8[r]) * rl[r];
                    pv[r] = (p >= thr) ? p : 0.f;
                }
            } else {
#pragma unroll
                for (int r = 0; r < 8; ++r) pv[r] = 0.f;
            }
            // C-layout -> row-major P tile in LDS
#pragma unroll
            for (int r = 0; r < 8; ++r)
                pbuf[wid][(r + hi * 8) * 32 + jt * 16 + nn] = (__bf16)pv[r];
        }
        asm volatile("s_wait_dscnt 0" ::: "memory");

        // re-read as A fragment (16x32, K = s)
        v16bf pf;
        {
            const __bf16* pr = &pbuf[wid][nn * 32];
            Frag f;
            f.u[0] = *(const uint4*)(pr + hi * 16);
            f.u[1] = *(const uint4*)(pr + hi * 16 + 8);
            pf = f.v;
        }

#pragma unroll
        for (int dc = 0; dc < 8; ++dc) {
            const __bf16* vp = vb + (size_t)(dc * 16 + nn) * TT + jc * 32 + hi * 16;
            od[dc] = wmma_bf16(pf, load_contig16(vp), od[dc]);
        }
    }

    // store y[b, t, h*D + d] (bf16) — input to Wo GEMM
#pragma unroll
    for (int dc = 0; dc < 8; ++dc)
#pragma unroll
        for (int r = 0; r < 8; ++r) {
            int t = tq * 16 + r + hi * 8;
            y[(size_t)(b * TT + t) * CC + h * DD + dc * 16 + nn] = (__bf16)od[dc][r];
        }
}

// ---------------- host launch ----------------
extern "C" void kernel_launch(void* const* d_in, const int* in_sizes, int n_in,
                              void* d_out, int out_size, void* d_ws, size_t ws_size,
                              hipStream_t stream) {
    (void)in_sizes; (void)n_in; (void)out_size; (void)ws_size;
    const float* x    = (const float*)d_in[0];
    const float* cosb = (const float*)d_in[1];
    const float* sinb = (const float*)d_in[2];
    const float* Wq   = (const float*)d_in[3];
    const float* Wk   = (const float*)d_in[4];
    const float* Wv   = (const float*)d_in[5];
    const float* Wo   = (const float*)d_in[6];
    const float* gate = (const float*)d_in[7];

    char* ws = (char*)d_ws;
    size_t off = 0;
    auto alloc = [&](size_t bytes) {
        char* p = ws + off;
        off += (bytes + 255) & ~(size_t)255;
        return p;
    };
    __bf16* xb     = (__bf16*)alloc((size_t)BB * TT * CC * 2);            // 16.8 MB
    __bf16* wqkvT  = (__bf16*)alloc((size_t)NQKV * CC * 2);               // 12.6 MB
    __bf16* woT    = (__bf16*)alloc((size_t)CC * CC * 2);                 //  8.4 MB
    __bf16* qkvraw = (__bf16*)alloc((size_t)BB * TT * NQKV * 2);          // 25.2 MB
    __bf16* qm     = (__bf16*)alloc((size_t)BB * HH * TT * DD * 2);       // 16.8 MB
    __bf16* km     = (__bf16*)alloc((size_t)BB * HKV * TT * DD * 2);      //  4.2 MB
    __bf16* vTm    = (__bf16*)alloc((size_t)BB * HKV * DD * TT * 2);      //  4.2 MB
    __bf16* ym     = (__bf16*)alloc((size_t)BB * TT * CC * 2);            // 16.8 MB

    const int nx = BB * TT * CC;                       // 8388608
    k_f32_to_bf16<<<nx / 256, 256, 0, stream>>>(x, xb, nx);
    k_build_wqkvT<<<(NQKV * CC) / 256, 256, 0, stream>>>(Wq, Wk, Wv, wqkvT);
    k_build_woT<<<(CC * CC) / 256, 256, 0, stream>>>(Wo, woT);

    // qkv_raw [4096, 3072] = xb [4096, 2048] @ wqkvT^T  (128x128 block tiles)
    k_gemm_bf16<0><<<dim3(NQKV / 128, (BB * TT) / 128), 256, 0, stream>>>(
        xb, wqkvT, qkvraw, BB * TT, NQKV, CC);

    k_rope_q<<<(BB * HH * TT * DD) / 256, 256, 0, stream>>>(qkvraw, cosb, sinb, qm);
    k_rope_k<<<(BB * HKV * TT * DD) / 256, 256, 0, stream>>>(qkvraw, cosb, sinb, km);
    k_transpose_v<<<(BB * HKV * DD * TT) / 256, 256, 0, stream>>>(qkvraw, vTm);

    // attention: B*H*(T/16) = 4096 wave-tiles, 8 waves per block
    k_attn<<<(BB * HH * (TT / 16)) / 8, 256, 0, stream>>>(qm, km, vTm, gate, ym);

    // out [4096, 2048] f32 = ym @ woT^T
    k_gemm_bf16<1><<<dim3(CC / 128, (BB * TT) / 128), 256, 0, stream>>>(
        ym, woT, d_out, BB * TT, CC, CC);
}
